// Up_NodeModel_34273839022281
// MI455X (gfx1250) — compile-verified, matrix operand order checked by reading, not run on hardware
//
#include <hip/hip_runtime.h>
#include <hip/hip_bf16.h>

#define NFINE   100000
#define NCOARSE 25000
#define HDIM    128
#define TOT     (2 * NFINE)          // B * N_FINE rows

typedef __attribute__((ext_vector_type(16))) __bf16 v16bf;
typedef __attribute__((ext_vector_type(8)))  __bf16 v8bf;
typedef __attribute__((ext_vector_type(8)))  float  v8f;
typedef __attribute__((ext_vector_type(4)))  float  f4;

__device__ __forceinline__ v8f wmma_bf16(v16bf a, v16bf b, v8f c) {
  // (neg_a, A, neg_b, B, c_mod, C, reuse_a, reuse_b)
  return __builtin_amdgcn_wmma_f32_16x16x32_bf16(false, a, false, b, (short)0, c,
                                                 false, false);
}

// Stage a [K x 128] fp32 weight matrix into LDS as bf16, pre-swizzled into the
// 16-bit B-fragment layout: fragment (kt,nt) = 32 lanes x 16 bf16 contiguous.
// Within a 32-wide K block: lane-half h, slot j  ->  K = (j<8) ? h*8+j : 16+h*8+(j-8)
__device__ __forceinline__ void stage_weights(const float* __restrict__ W,
                                              __bf16* __restrict__ dst,
                                              int K, int tid) {
  for (int idx = tid; idx < K * HDIM; idx += 256) {
    int k = idx >> 7, n = idx & 127;
    int kt = k >> 5, kk = k & 31;
    int h, j;
    if (kk < 8)       { h = 0; j = kk;      }
    else if (kk < 16) { h = 1; j = kk - 8;  }
    else if (kk < 24) { h = 0; j = kk - 8;  }
    else              { h = 1; j = kk - 16; }
    int nt = n >> 4, nn = n & 15;
    int lane = h * 16 + nn;
    dst[(kt * 8 + nt) * 512 + lane * 16 + j] = (__bf16)W[idx];
  }
}

// Hidden layer (K=128): A fragments read from per-wave LDS scratch (row-major
// bf16 16x128), B fragments from swizzled LDS weights. 32 WMMAs.
__device__ __forceinline__ void hidden_layer(const __bf16* __restrict__ scr,
                                             const __bf16* __restrict__ wsw,
                                             int lane, v8f acc[8]) {
  int half = lane >> 4, mlo = lane & 15;
#pragma unroll
  for (int nt = 0; nt < 8; ++nt)
#pragma unroll
    for (int r = 0; r < 8; ++r) acc[nt][r] = 0.f;

#pragma unroll
  for (int kt = 0; kt < 4; ++kt) {
    const __bf16* ap = scr + mlo * HDIM + kt * 32 + half * 8;
    v8bf lo = *(const v8bf*)(ap);          // K = kt*32 + h*8 .. +7
    v8bf hi = *(const v8bf*)(ap + 16);     // K = kt*32 + 16 + h*8 .. +7
    v16bf a;
#pragma unroll
    for (int q = 0; q < 8; ++q) { a[q] = lo[q]; a[8 + q] = hi[q]; }
#pragma unroll
    for (int nt = 0; nt < 8; ++nt) {
      v16bf bf = *(const v16bf*)(wsw + (kt * 8 + nt) * 512 + lane * 16);
      acc[nt] = wmma_bf16(a, bf, acc[nt]);
    }
  }
}

extern "C" __global__ void __launch_bounds__(256, 2)
node_mlp_kernel(const float* __restrict__ x, const float* __restrict__ xs,
                const int* __restrict__ f2c, const float* __restrict__ dist,
                const float* __restrict__ W1, const float* __restrict__ b1,
                const float* __restrict__ W2, const float* __restrict__ b2,
                const float* __restrict__ W3, const float* __restrict__ b3,
                float* __restrict__ out) {
  extern __shared__ char smem[];
  __bf16* w1s = (__bf16*)smem;                    //  64 KB: 8 kt x 8 nt frags
  __bf16* w2s = (__bf16*)(smem + 65536);          //  32 KB
  __bf16* w3s = (__bf16*)(smem + 98304);          //  32 KB
  __bf16* scrb = (__bf16*)(smem + 131072);        //  32 KB: 8 waves x 16x128

  const int tid  = threadIdx.x;
  const int lane = tid & 31, wave = tid >> 5;
  const int half = lane >> 4, mlo = lane & 15;
  __bf16* scr = scrb + wave * (16 * HDIM);

  stage_weights(W1, w1s, 256, tid);
  stage_weights(W2, w2s, 128, tid);
  stage_weights(W3, w3s, 128, tid);
  __syncthreads();

  const int r0 = blockIdx.x * 128 + wave * 16;

  // ---- layer 1: A = [gather(x) | x_scale], K = 256 ----
  // A layout: lane&15 = row M, lane-half selects K sub-groups.
  int gA  = r0 + mlo;
  int gAc = gA < TOT ? gA : TOT - 1;
  int bb  = gAc / NFINE, ii = gAc - bb * NFINE;
  int ci  = f2c[ii];
  const float* xrow  = x  + ((size_t)bb * NCOARSE + ci) * HDIM;
  const float* xsrow = xs + (size_t)gAc * HDIM;

  // distances per C-row (row m = half*8 + r)
  float dvr[8];
#pragma unroll
  for (int r = 0; r < 8; ++r) {
    int g = r0 + half * 8 + r;
    g = g < TOT ? g : TOT - 1;
    dvr[r] = dist[g % NFINE];
  }

  v8f acc[8];
#pragma unroll
  for (int nt = 0; nt < 8; ++nt)
#pragma unroll
    for (int r = 0; r < 8; ++r) acc[nt][r] = 0.f;

#pragma unroll
  for (int kt = 0; kt < 8; ++kt) {
    const float* src = (kt < 4) ? (xrow + kt * 32) : (xsrow + (kt - 4) * 32);
    const int cb = half * 8;
    f4 f0 = *(const f4*)(src + cb);
    f4 f1 = *(const f4*)(src + cb + 4);
    f4 f2 = *(const f4*)(src + cb + 16);
    f4 f3 = *(const f4*)(src + cb + 20);
    v16bf a;
#pragma unroll
    for (int q = 0; q < 4; ++q) {
      a[q]      = (__bf16)f0[q];
      a[4 + q]  = (__bf16)f1[q];
      a[8 + q]  = (__bf16)f2[q];
      a[12 + q] = (__bf16)f3[q];
    }
#pragma unroll
    for (int nt = 0; nt < 8; ++nt) {
      v16bf bf = *(const v16bf*)(w1s + (kt * 8 + nt) * 512 + lane * 16);
      acc[nt] = wmma_bf16(a, bf, acc[nt]);
    }
  }

  // epilogue 1: + b1 + d * W1[256,:]  ; SiLU ; -> scratch as bf16
#pragma unroll
  for (int nt = 0; nt < 8; ++nt) {
    int n = nt * 16 + mlo;
    float wl = W1[256 * HDIM + n];
    float bs = b1[n];
#pragma unroll
    for (int r = 0; r < 8; ++r) {
      float v = acc[nt][r] + dvr[r] * wl + bs;
      v = v / (1.f + __expf(-v));
      scr[(half * 8 + r) * HDIM + n] = (__bf16)v;
    }
  }
  __syncthreads();

  // ---- layer 2 ----
  hidden_layer(scr, w2s, lane, acc);
#pragma unroll
  for (int nt = 0; nt < 8; ++nt) {
    int n = nt * 16 + mlo;
    float bs = b2[n];
#pragma unroll
    for (int r = 0; r < 8; ++r) {
      float v = acc[nt][r] + bs;
      v = v / (1.f + __expf(-v));
      scr[(half * 8 + r) * HDIM + n] = (__bf16)v;
    }
  }
  __syncthreads();

  // ---- layer 3 ----
  hidden_layer(scr, w3s, lane, acc);
#pragma unroll
  for (int nt = 0; nt < 8; ++nt) {
    int n = nt * 16 + mlo;
    float bs = b3[n];
#pragma unroll
    for (int r = 0; r < 8; ++r) {
      int g = r0 + half * 8 + r;
      if (g < TOT) out[(size_t)g * HDIM + n] = acc[nt][r] + bs;
    }
  }
}

extern "C" void kernel_launch(void* const* d_in, const int* in_sizes, int n_in,
                              void* d_out, int out_size, void* d_ws, size_t ws_size,
                              hipStream_t stream) {
  const float* x    = (const float*)d_in[0];
  const float* xs   = (const float*)d_in[1];
  const int*   f2c  = (const int*)d_in[2];
  const float* dist = (const float*)d_in[3];
  const float* W1   = (const float*)d_in[4];
  const float* b1   = (const float*)d_in[5];
  const float* W2   = (const float*)d_in[6];
  const float* b2   = (const float*)d_in[7];
  const float* W3   = (const float*)d_in[8];
  const float* b3   = (const float*)d_in[9];
  float* out = (float*)d_out;

  const size_t smem = 160 * 1024;   // 2 blocks/WGP of 320 KB LDS
  hipFuncSetAttribute((const void*)node_mlp_kernel,
                      hipFuncAttributeMaxDynamicSharedMemorySize, (int)smem);
  int grid = (TOT + 127) / 128;     // 128 rows per block
  node_mlp_kernel<<<grid, 256, smem, stream>>>(x, xs, f2c, dist,
                                               W1, b1, W2, b2, W3, b3, out);
}
